// TransformerGNN_66975720014405
// MI455X (gfx1250) — matrix operand order repaired and checked by reading, not compiled, not run on hardware
//
#include <hip/hip_runtime.h>

// ---------------------------------------------------------------------------
// Types for CDNA5 WMMA (wave32): bf16 16x16x32, f32 accumulate
// ---------------------------------------------------------------------------
typedef __bf16 bf16_t;
typedef bf16_t v16bf __attribute__((ext_vector_type(16)));
typedef bf16_t v8bf  __attribute__((ext_vector_type(8)));
typedef bf16_t v2bf  __attribute__((ext_vector_type(2)));
typedef float  v8f   __attribute__((ext_vector_type(8)));

typedef unsigned int u32x4 __attribute__((ext_vector_type(4)));
typedef int          i32x4 __attribute__((ext_vector_type(4)));
typedef int          i32x8 __attribute__((ext_vector_type(8)));

#define SST 72   // LDS row stride (bf16 elems): 144 B = 9*16 -> 16B-aligned runs

// ---------------------------------------------------------------------------
// A fragment: 16x32 bf16 from row-major LDS (stride SST). Lane L -> row
// M = L&15, hi = L>>4; elems 0..7 -> K = 8*hi..8*hi+7, elems 8..15 ->
// K = 16+8*hi.. (ISA 7.12.2). Both runs are contiguous + 16B aligned.
// ---------------------------------------------------------------------------
static __device__ __forceinline__ v16bf load_a_frag(const bf16_t* sh, int mbase,
                                                    int kbase, int lane) {
  int m  = mbase + (lane & 15);
  int hi = lane >> 4;
  v8bf lo = *(const v8bf*)(sh + m * SST + kbase + 8 * hi);
  v8bf hv = *(const v8bf*)(sh + m * SST + kbase + 16 + 8 * hi);
  return __builtin_shufflevector(lo, hv, 0, 1, 2, 3, 4, 5, 6, 7, 8, 9, 10, 11,
                                 12, 13, 14, 15);
}

// ---------------------------------------------------------------------------
// B fragments come from a pre-swizzled LDS buffer shWf[4][2][32][16]:
// blk = nt*2+kb holds, for each lane, its 16 fragment elements contiguously:
// shWf[((nt*2+kb)*32 + L)*16 + e] = W[(kb*32 + 8*(L>>4) + e + 8*(e>=8))*64 + nt*16+(L&15)]
// -> one aligned 32-byte v16bf load per fragment.
// ---------------------------------------------------------------------------
static __device__ __forceinline__ v16bf load_b_frag(const bf16_t* shWf, int nt,
                                                    int kbase, int lane) {
  int blk = nt * 2 + (kbase >> 5);
  return *(const v16bf*)(shWf + (blk * 32 + lane) * 16);
}

static __device__ __forceinline__ void gemm_tile_64(const bf16_t* shIn,
                                                    const bf16_t* shWf, int mt,
                                                    int nt, int lane, v8f& c) {
  v16bf a0 = load_a_frag(shIn, mt * 16, 0, lane);
  v16bf b0 = load_b_frag(shWf, nt, 0, lane);
  c = __builtin_amdgcn_wmma_f32_16x16x32_bf16(false, a0, false, b0, (short)0, c,
                                              false, false);
  v16bf a1 = load_a_frag(shIn, mt * 16, 32, lane);
  v16bf b1 = load_b_frag(shWf, nt, 32, lane);
  c = __builtin_amdgcn_wmma_f32_16x16x32_bf16(false, a1, false, b1, (short)0, c,
                                              false, false);
}

// Swizzle a 64x64 f32 weight (row-major, arbitrary load functor src(k,n))
// into fragment-order bf16 LDS.
template <typename F>
static __device__ __forceinline__ void stage_w_swizzled(F src, bf16_t* shWf,
                                                        int tid, int nthr) {
  for (int idx = tid; idx < 4096; idx += nthr) {
    int blk = idx >> 9, rem = idx & 511;
    int L = rem >> 4, e = rem & 15;
    int nt = blk >> 1, kb = blk & 1;
    int k = kb * 32 + 8 * (L >> 4) + e + (e >= 8 ? 8 : 0);
    int n = nt * 16 + (L & 15);
    shWf[idx] = (bf16_t)src(k, n);
  }
}

// ---------------------------------------------------------------------------
// TDM: async 64x64 f32 tile load, global -> LDS (D# per ISA 08 sec.8).
// group0: count=1 | lds_addr | global_addr[56:0] | type=2
// group1: data_size=4B; tensor_dim0=64, tensor_dim1=64; tile 64x64; stride0=64
// Toolchain here uses the 6-arg builtin: (g0, g1, g2, g3, g4, cpol).
// ---------------------------------------------------------------------------
static __device__ __forceinline__ void tdm_load_64x64_f32(const float* gptr,
                                                          unsigned lds_off) {
  unsigned long long ga = (unsigned long long)(size_t)gptr;
  u32x4 g0 = {1u, lds_off, (unsigned)ga,
              (unsigned)((ga >> 32) & 0x01FFFFFFu) | 0x80000000u};
  i32x8 g1 = {(int)0x00020000,      // data_size = 4B
              (int)(64u << 16),     // tensor_dim0 = 64 (lo16 @ bits 63:48)
              (int)(64u << 16),     // dim0 hi=0 | tensor_dim1 lo16 = 64
              (int)(64u << 16),     // dim1 hi=0 | tile_dim0 = 64
              64,                   // tile_dim1 = 64, tile_dim2 = 0
              64,                   // tensor_dim0_stride = 64 (lo32)
              0, 0};
  i32x4 gz4 = {0, 0, 0, 0};
  i32x8 gz8 = {0, 0, 0, 0, 0, 0, 0, 0};
  __builtin_amdgcn_tensor_load_to_lds(g0, g1, gz4, gz4, gz8, 0);
}

// ---------------------------------------------------------------------------
// KNN stage 1: partial top-16 per (point, segment). Grid (N/256, KSEG).
// ---------------------------------------------------------------------------
#define KSEG 8

__global__ __launch_bounds__(256) void knn_part_kernel(
    const float* __restrict__ pos, float* __restrict__ partd,
    int* __restrict__ parti, int N) {
  __shared__ float sp[256 * 3];
  int tid = threadIdx.x;
  int i   = blockIdx.x * 256 + tid;
  int seg = blockIdx.y;
  int j0 = seg * (N / KSEG), j1 = j0 + (N / KSEG);

  float px = pos[i * 3 + 0], py = pos[i * 3 + 1], pz = pos[i * 3 + 2];

  float bd[16];
  int   bi[16];
#pragma unroll
  for (int k = 0; k < 16; ++k) { bd[k] = 3.4e38f; bi[k] = 0; }
  float wmax = 3.4e38f;

  for (int jt = j0; jt < j1; jt += 256) {
    for (int q = tid; q < 256 * 3; q += 256) sp[q] = pos[jt * 3 + q];
    if (jt + 256 < j1) __builtin_prefetch(&pos[(jt + 256) * 3], 0, 1);
    __syncthreads();
#pragma unroll 4
    for (int jj = 0; jj < 256; ++jj) {
      int   j  = jt + jj;
      float dx = px - sp[jj * 3 + 0];
      float dy = py - sp[jj * 3 + 1];
      float dz = pz - sp[jj * 3 + 2];
      float d  = dx * dx + dy * dy + dz * dz;
      if (j != i && d < wmax) {
        int   am = 0;
        float mv = bd[0];
#pragma unroll
        for (int k = 1; k < 16; ++k)
          if (bd[k] > mv) { mv = bd[k]; am = k; }
#pragma unroll
        for (int k = 0; k < 16; ++k)
          if (k == am) { bd[k] = d; bi[k] = j; }
        wmax = bd[0];
#pragma unroll
        for (int k = 1; k < 16; ++k) wmax = fmaxf(wmax, bd[k]);
      }
    }
    __syncthreads();
  }
  size_t base = ((size_t)i * KSEG + seg) * 16;
#pragma unroll
  for (int k = 0; k < 16; ++k) {
    partd[base + k] = bd[k];
    parti[base + k] = bi[k];
  }
}

// KNN stage 2: merge KSEG*16 partial candidates -> global top-16 per point.
__global__ __launch_bounds__(256) void knn_merge_kernel(
    const float* __restrict__ partd, const int* __restrict__ parti,
    int* __restrict__ knn) {
  int i = blockIdx.x * 256 + threadIdx.x;
  float bd[16];
  int   bi[16];
#pragma unroll
  for (int k = 0; k < 16; ++k) { bd[k] = 3.4e38f; bi[k] = 0; }
  float wmax = 3.4e38f;
  size_t base = (size_t)i * KSEG * 16;
  for (int q = 0; q < KSEG * 16; ++q) {
    float d = partd[base + q];
    int   j = parti[base + q];
    if (d < wmax) {
      int   am = 0;
      float mv = bd[0];
#pragma unroll
      for (int k = 1; k < 16; ++k)
        if (bd[k] > mv) { mv = bd[k]; am = k; }
#pragma unroll
      for (int k = 0; k < 16; ++k)
        if (k == am) { bd[k] = d; bi[k] = j; }
      wmax = bd[0];
#pragma unroll
      for (int k = 1; k < 16; ++k) wmax = fmaxf(wmax, bd[k]);
    }
  }
#pragma unroll
  for (int k = 0; k < 16; ++k) knn[i * 16 + k] = bi[k];
}

// ---------------------------------------------------------------------------
// Dense [M,64] @ [64,64] + bias (+ residual) via WMMA. A-slab and W staged
// global->LDS by the Tensor Data Mover, then converted/swizzled to bf16.
// Block = 128 threads (4 waves) = 64-row slab; 16 tiles, 4 per wave.
// ---------------------------------------------------------------------------
template <bool BIAS, bool RESID>
__global__ __launch_bounds__(128) void gemm64_kernel(
    const float* __restrict__ A, const float* __restrict__ W,
    const float* __restrict__ bias, const float* __restrict__ resid,
    float* __restrict__ out) {
  __shared__ __align__(16) float  stageA[64 * 64];
  __shared__ __align__(16) float  stageW[64 * 64];
  __shared__ __align__(16) bf16_t shA[64 * SST];
  __shared__ __align__(32) bf16_t shWf[4096];

  int tid = threadIdx.x, lane = tid & 31, wave = tid >> 5;
  int r0  = blockIdx.x * 64;

  if (wave == 0) {   // one wave issues both TDM transfers, fences its counter
    tdm_load_64x64_f32(A + (size_t)r0 * 64, (unsigned)(size_t)(void*)stageA);
    tdm_load_64x64_f32(W, (unsigned)(size_t)(void*)stageW);
    __builtin_amdgcn_s_wait_tensorcnt(0);
  }
  __syncthreads();

  // convert f32 slab -> bf16 (row stride SST), weight -> fragment-swizzled
  for (int i = tid * 2; i < 4096; i += 256) {
    v2bf t = {(bf16_t)stageA[i], (bf16_t)stageA[i + 1]};
    *(v2bf*)&shA[(i >> 6) * SST + (i & 63)] = t;
  }
  stage_w_swizzled([&](int k, int n) { return stageW[k * 64 + n]; }, shWf, tid,
                   128);
  __syncthreads();

  for (int t = 0; t < 4; ++t) {
    int tile = t * 4 + wave;
    int mt = tile >> 2, nt = tile & 3;
    v8f c = {};
    gemm_tile_64(shA, shWf, mt, nt, lane, c);

    int   n  = nt * 16 + (lane & 15);
    int   hi = lane >> 4;
    float b  = BIAS ? bias[n] : 0.0f;
#pragma unroll
    for (int r = 0; r < 8; ++r) {
      int   row = r0 + mt * 16 + r + 8 * hi;
      float o   = c[r] + b;
      if (RESID) o += resid[row * 64 + n];
      out[row * 64 + n] = o;
    }
  }
}

// ---------------------------------------------------------------------------
// Fused per-edge pipeline: 8 points x 16 neighbors = 128 edge rows in LDS.
// Three WMMA GEMMs; softmax over k fused into the last GEMM's register
// epilogue (k == M dim of each 16x16 tile -> 8-reg reduce + shfl_xor(16)).
// ---------------------------------------------------------------------------
#define EP 8
#define EE (EP * 16)

__global__ __launch_bounds__(128) void edge_kernel(
    const float* __restrict__ pos, const int* __restrict__ knn,
    const float* __restrict__ a_dst, const float* __restrict__ a_src,
    const float* __restrict__ vmat,
    const float* __restrict__ p1w, const float* __restrict__ p1b,
    const float* __restrict__ p1g, const float* __restrict__ p1bt,
    const float* __restrict__ p2w, const float* __restrict__ p2b,
    const float* __restrict__ a1w, const float* __restrict__ a1b,
    const float* __restrict__ a1g, const float* __restrict__ a1bt,
    const float* __restrict__ a2w, const float* __restrict__ a2b,
    float* __restrict__ out_pre) {
  __shared__ __align__(16) bf16_t shA[EE * SST];  // GEMM input (h1 / alpha0)
  __shared__ __align__(16) bf16_t shH[EE * SST];  // attn hidden
  __shared__ __align__(16) bf16_t shD[EE * SST];  // delta, reused at the end
  __shared__ __align__(32) bf16_t shWf[4096];     // fragment-swizzled weight
  __shared__ int sidx[EE];

  int tid = threadIdx.x, lane = tid & 31, wave = tid >> 5;
  int p0  = blockIdx.x * EP;

  // ---- stage ids, p2w, and h1 = relu(bn(rel @ p1w + b)) ----
  sidx[tid] = knn[p0 * 16 + tid];
  stage_w_swizzled([&](int k, int n) { return p2w[k * 64 + n]; }, shWf, tid, 128);
  {
    int e  = tid;
    int pi = p0 + (e >> 4);
    int j  = sidx[e];  // own write
    float rx = pos[pi * 3 + 0] - pos[j * 3 + 0];
    float ry = pos[pi * 3 + 1] - pos[j * 3 + 1];
    float rz = pos[pi * 3 + 2] - pos[j * 3 + 2];
    for (int c = 0; c < 64; c += 2) {
      float h0 = rx * p1w[c] + ry * p1w[64 + c] + rz * p1w[128 + c] + p1b[c];
      float h1 = rx * p1w[c + 1] + ry * p1w[64 + c + 1] + rz * p1w[128 + c + 1] +
                 p1b[c + 1];
      h0 = fmaxf(p1g[c] * h0 + p1bt[c], 0.0f);
      h1 = fmaxf(p1g[c + 1] * h1 + p1bt[c + 1], 0.0f);
      v2bf t = {(bf16_t)h0, (bf16_t)h1};
      *(v2bf*)&shA[e * SST + c] = t;
    }
  }
  __syncthreads();

  // ---- GEMM: delta = h1 @ p2w + p2b -> shD ----
  for (int t = 0; t < 8; ++t) {
    int tile = t * 4 + wave;
    int mt = tile >> 2, nt = tile & 3;
    v8f c = {};
    gemm_tile_64(shA, shWf, mt, nt, lane, c);
    int   n  = nt * 16 + (lane & 15);
    int   hi = lane >> 4;
    float b  = p2b[n];
#pragma unroll
    for (int r = 0; r < 8; ++r)
      shD[(mt * 16 + r + 8 * hi) * SST + n] = (bf16_t)(c[r] + b);
  }
  __syncthreads();

  // ---- alpha0 = a_dst[i] - a_src[j] + delta -> shA ; stage a1w ----
  {
    int e  = tid;
    int pi = p0 + (e >> 4);
    int j  = sidx[e];
    const float2* pd = (const float2*)(a_dst + (size_t)pi * 64);
    const float2* ps = (const float2*)(a_src + (size_t)j * 64);
    for (int c = 0; c < 64; c += 2) {
      float2 d2 = pd[c >> 1], s2 = ps[c >> 1];
      v2bf dd = *(const v2bf*)&shD[e * SST + c];
      v2bf t = {(bf16_t)(d2.x - s2.x + (float)dd[0]),
                (bf16_t)(d2.y - s2.y + (float)dd[1])};
      *(v2bf*)&shA[e * SST + c] = t;
    }
  }
  stage_w_swizzled([&](int k, int n) { return a1w[k * 64 + n]; }, shWf, tid, 128);
  __syncthreads();

  // ---- GEMM: h2 = relu(bn(alpha0 @ a1w + a1b)) -> shH ----
  for (int t = 0; t < 8; ++t) {
    int tile = t * 4 + wave;
    int mt = tile >> 2, nt = tile & 3;
    v8f c = {};
    gemm_tile_64(shA, shWf, mt, nt, lane, c);
    int   n  = nt * 16 + (lane & 15);
    int   hi = lane >> 4;
    float b = a1b[n], g = a1g[n], bt = a1bt[n];
#pragma unroll
    for (int r = 0; r < 8; ++r)
      shH[(mt * 16 + r + 8 * hi) * SST + n] =
          (bf16_t)fmaxf(g * (c[r] + b) + bt, 0.0f);
  }
  __syncthreads();

  stage_w_swizzled([&](int k, int n) { return a2w[k * 64 + n]; }, shWf, tid, 128);
  __syncthreads();

  // ---- GEMM: alpha = h2 @ a2w + a2b, fused softmax(k) + weighted sum ----
  for (int t = 0; t < 8; ++t) {
    int tile = t * 4 + wave;
    int mt = tile >> 2, nt = tile & 3;
    v8f c = {};
    gemm_tile_64(shH, shWf, mt, nt, lane, c);

    int   n  = nt * 16 + (lane & 15);
    int   hi = lane >> 4;
    float b  = a2b[n];

    float al[8];
#pragma unroll
    for (int r = 0; r < 8; ++r) al[r] = c[r] + b;

    float m = al[0];
#pragma unroll
    for (int r = 1; r < 8; ++r) m = fmaxf(m, al[r]);
    m = fmaxf(m, __shfl_xor(m, 16, 32));  // join M=0..7 with M=8..15

    float ex[8];
    float s = 0.0f;
#pragma unroll
    for (int r = 0; r < 8; ++r) { ex[r] = __expf(al[r] - m); s += ex[r]; }
    s += __shfl_xor(s, 16, 32);
    float inv = 1.0f / s;

    int   pi  = p0 + mt;
    float acc = 0.0f;
#pragma unroll
    for (int r = 0; r < 8; ++r) {
      int   k   = r + 8 * hi;
      int   j   = sidx[mt * 16 + k];
      float val = vmat[(size_t)j * 64 + n] + (float)shD[(mt * 16 + k) * SST + n];
      acc += ex[r] * inv * val;
    }
    acc += __shfl_xor(acc, 16, 32);
    if (lane < 16) out_pre[(size_t)pi * 64 + n] = acc;
  }
}

// ---------------------------------------------------------------------------
// Host-side orchestration
// ---------------------------------------------------------------------------
extern "C" void kernel_launch(void* const* d_in, const int* in_sizes, int n_in,
                              void* d_out, int out_size, void* d_ws,
                              size_t ws_size, hipStream_t stream) {
  const int N = 8192, C = 64;

  const float* x     = (const float*)d_in[0];
  const float* pos   = (const float*)d_in[1];
  const float* w_src = (const float*)d_in[2];
  const float* w_dst = (const float*)d_in[3];
  const float* w_lin = (const float*)d_in[4];
  const float* b_lin = (const float*)d_in[5];
  const float* p1w   = (const float*)d_in[6];
  const float* p1b   = (const float*)d_in[7];
  const float* p1g   = (const float*)d_in[8];
  const float* p1bt  = (const float*)d_in[9];
  const float* p2w   = (const float*)d_in[10];
  const float* p2b   = (const float*)d_in[11];
  const float* a1w   = (const float*)d_in[12];
  const float* a1b   = (const float*)d_in[13];
  const float* a1g   = (const float*)d_in[14];
  const float* a1bt  = (const float*)d_in[15];
  const float* a2w   = (const float*)d_in[16];
  const float* a2b   = (const float*)d_in[17];
  const float* uw    = (const float*)d_in[18];
  const float* ub    = (const float*)d_in[19];
  float*       out   = (float*)d_out;

  // Workspace carve-up (all fully rewritten every call).
  char*  ws    = (char*)d_ws;
  int*   knn   = (int*)ws;                         // N*16 ints      (512 KB)
  float* partd = (float*)(ws + (size_t)N * 16 * 4);        // N*KSEG*16 (4 MB)
  int*   parti = (int*)((char*)partd + (size_t)N * KSEG * 16 * 4);  // (4 MB)
  float* adst  = (float*)((char*)parti + (size_t)N * KSEG * 16 * 4);
  float* asrc  = adst + (size_t)N * C;
  float* vv    = asrc + (size_t)N * C;
  float* opre  = vv + (size_t)N * C;

  dim3 kgrid(N / 256, KSEG);
  knn_part_kernel<<<kgrid, 256, 0, stream>>>(pos, partd, parti, N);
  knn_merge_kernel<<<N / 256, 256, 0, stream>>>(partd, parti, knn);

  gemm64_kernel<false, false><<<N / 64, 128, 0, stream>>>(x, w_dst, nullptr,
                                                          nullptr, adst);
  gemm64_kernel<false, false><<<N / 64, 128, 0, stream>>>(x, w_src, nullptr,
                                                          nullptr, asrc);
  gemm64_kernel<true, false><<<N / 64, 128, 0, stream>>>(x, w_lin, b_lin,
                                                         nullptr, vv);

  edge_kernel<<<N / EP, 128, 0, stream>>>(pos, knn, adst, asrc, vv, p1w, p1b,
                                          p1g, p1bt, p2w, p2b, a1w, a1b, a1g,
                                          a1bt, a2w, a2b, opre);

  // out = out_pre @ uw + ub + x  (residual fused)
  gemm64_kernel<true, true><<<N / 64, 128, 0, stream>>>(opre, uw, ub, x, out);
}
